// RetNetBlock_70171175682721
// MI455X (gfx1250) — compile-verified, matrix operand order checked by reading, not run on hardware
//
#include <hip/hip_runtime.h>
#include <cstdint>
#include <cstddef>

#define BATCH 8
#define SEQ   256
#define DM    512
#define NH    8
#define HD    64
#define TOK   (BATCH * SEQ)   // 2048
#define FFNS  1024
#define FLATD (SEQ * DM)      // 131072

typedef __attribute__((ext_vector_type(16))) __bf16 v16bf;
typedef __attribute__((ext_vector_type(8)))  float  v8f;

union FragU { v16bf v; unsigned short u[16]; uint4 q[2]; };

__device__ __forceinline__ unsigned short f2bf(float f) {
  unsigned x = __float_as_uint(f);
  x += 0x7FFFu + ((x >> 16) & 1u);        // round-to-nearest-even
  return (unsigned short)(x >> 16);
}
__device__ __forceinline__ float bf2f(unsigned short h) {
  return __uint_as_float(((unsigned)h) << 16);
}

// A/B fragment, K-contiguous source (row-major [rows, ld]):
// lane holds row (lane&15); element e -> k = k0 + (e<8?0:16) + (lane>>4)*8 + (e&7)
// => two 16B loads per lane.
__device__ __forceinline__ v16bf load_frag_k(const unsigned short* __restrict__ base,
                                             int ld, int row0, int k0) {
  const int lane = threadIdx.x & 31;
  const int row  = row0 + (lane & 15);
  const int half = lane >> 4;
  const unsigned short* p = base + (size_t)row * ld + k0 + half * 8;
  FragU u;
  u.q[0] = *reinterpret_cast<const uint4*>(p);
  u.q[1] = *reinterpret_cast<const uint4*>(p + 16);
  return u.v;
}

// B fragment from row-major [K, N] (column gather, used only for V).
__device__ __forceinline__ v16bf load_frag_strided(const unsigned short* __restrict__ base,
                                                   int ld, int k0, int col0) {
  const int lane = threadIdx.x & 31;
  const int col  = col0 + (lane & 15);
  const int half = lane >> 4;
  FragU u;
#pragma unroll
  for (int e = 0; e < 16; ++e) {
    int k = k0 + ((e < 8) ? 0 : 16) + half * 8 + (e & 7);
    u.u[e] = base[(size_t)k * ld + col];
  }
  return u.v;
}

// ---------------- LayerNorm: one row (512) per 256-thread block ----------------
__global__ void k_layernorm(const float* __restrict__ x, const float* __restrict__ w,
                            const float* __restrict__ b,
                            unsigned short* __restrict__ obf, float* __restrict__ of32) {
  const int row = blockIdx.x, tid = threadIdx.x;
  const float* xr = x + (size_t)row * DM;
  float x0 = xr[tid], x1 = xr[tid + 256];
  __shared__ float ssum[256], ssq[256];
  ssum[tid] = x0 + x1;
  ssq[tid]  = x0 * x0 + x1 * x1;
  __syncthreads();
  for (int off = 128; off > 0; off >>= 1) {
    if (tid < off) { ssum[tid] += ssum[tid + off]; ssq[tid] += ssq[tid + off]; }
    __syncthreads();
  }
  const float mean = ssum[0] * (1.0f / DM);
  const float var  = ssq[0] * (1.0f / DM) - mean * mean;
  const float inv  = rsqrtf(var + 1e-5f);
  float y0 = (x0 - mean) * inv * w[tid]       + b[tid];
  float y1 = (x1 - mean) * inv * w[tid + 256] + b[tid + 256];
  size_t o = (size_t)row * DM + tid;
  obf[o] = f2bf(y0);  obf[o + 256] = f2bf(y1);
  of32[o] = y0;       of32[o + 256] = y1;
}

// ------------- Weight convert f32 [K,N] -> bf16 transposed [N,K] ---------------
__global__ void k_transpose_bf16(const float* __restrict__ W, unsigned short* __restrict__ Wt) {
  const int o = blockIdx.x * 256 + threadIdx.x;   // o = n*512 + k
  const int n = o >> 9, kk = o & 511;
  Wt[o] = f2bf(W[(size_t)kk * DM + n]);
}

// ---- WMMA GEMM: C[M,N] = A[M,K] * Bt[N,K]^T ; one 16x16 tile per wave ---------
// Cbf != null: store bf16. else: store f32 with residual add.
__global__ void k_gemm_bf16(const unsigned short* __restrict__ A,
                            const unsigned short* __restrict__ Bt,
                            unsigned short* __restrict__ Cbf,
                            float* __restrict__ Cf, const float* __restrict__ res,
                            int M, int N, int K) {
  const int wave = blockIdx.x * (blockDim.x >> 5) + (threadIdx.x >> 5);
  const int tilesN = N >> 4;
  if (wave >= (M >> 4) * tilesN) return;
  const int tm = (wave / tilesN) << 4;
  const int tn = (wave % tilesN) << 4;
  v8f acc = {};
  for (int k0 = 0; k0 < K; k0 += 32) {
    v16bf a = load_frag_k(A,  K, tm, k0);
    v16bf bfr = load_frag_k(Bt, K, tn, k0);
    acc = __builtin_amdgcn_wmma_f32_16x16x32_bf16(false, a, false, bfr,
                                                  (short)0, acc, false, false);
  }
  const int lane = threadIdx.x & 31, half = lane >> 4;
  const int n = tn + (lane & 15);
#pragma unroll
  for (int r = 0; r < 8; ++r) {
    const int m = tm + r + 8 * half;
    const size_t o = (size_t)m * N + n;
    if (Cbf) Cbf[o] = f2bf(acc[r]);
    else     Cf[o]  = acc[r] + res[o];
  }
}

// ------ Retention scores: S = (Q K^T / 8) * decay(n,m), stored bf16 ------------
__global__ void k_scores(const unsigned short* __restrict__ q,
                         const unsigned short* __restrict__ kmat,
                         unsigned short* __restrict__ scores) {
  const int wave = blockIdx.x * 8 + (threadIdx.x >> 5);
  if (wave >= BATCH * NH * 16 * 16) return;
  const int bh = wave >> 8, t = wave & 255;
  const int b = bh >> 3, hh = bh & 7;
  const int tn = (t >> 4) << 4;        // query rows
  const int tmk = (t & 15) << 4;       // key rows (score cols)
  const int col0 = hh * HD;
  v8f acc = {};
  for (int k0 = 0; k0 < HD; k0 += 32) {
    v16bf a = load_frag_k(q,    DM, b * SEQ + tn,  col0 + k0);
    v16bf bfr = load_frag_k(kmat, DM, b * SEQ + tmk, col0 + k0);
    acc = __builtin_amdgcn_wmma_f32_16x16x32_bf16(false, a, false, bfr,
                                                  (short)0, acc, false, false);
  }
  const float gamma = 1.0f - exp2f(-5.0f - (float)hh);
  const float l2g = __log2f(gamma);
  const int lane = threadIdx.x & 31, half = lane >> 4;
  unsigned short* srow = scores + (size_t)bh * SEQ * SEQ;
  const int m = tmk + (lane & 15);
#pragma unroll
  for (int r = 0; r < 8; ++r) {
    const int n = tn + r + 8 * half;
    const int d = n - m;
    float val = (d >= 0) ? acc[r] * 0.125f * exp2f((float)d * l2g) : 0.0f;
    srow[(size_t)n * SEQ + m] = f2bf(val);
  }
}

// --------------- Retention output: R = S_bf16 @ V ------------------------------
__global__ void k_retout(const unsigned short* __restrict__ scores,
                         const unsigned short* __restrict__ v,
                         unsigned short* __restrict__ ret) {
  const int wave = blockIdx.x * 8 + (threadIdx.x >> 5);
  if (wave >= BATCH * NH * 16 * 4) return;
  const int bh = wave >> 6, t = wave & 63;
  const int b = bh >> 3, hh = bh & 7;
  const int tn = (t >> 2) << 4;        // token rows
  const int td = (t & 3) << 4;         // head-dim cols
  const unsigned short* srow = scores + (size_t)bh * SEQ * SEQ;
  const unsigned short* vb = v + (size_t)b * SEQ * DM;
  v8f acc = {};
  for (int k0 = 0; k0 < SEQ; k0 += 32) {
    v16bf a = load_frag_k(srow, SEQ, tn, k0);
    v16bf bfr = load_frag_strided(vb, DM, k0, hh * HD + td);
    acc = __builtin_amdgcn_wmma_f32_16x16x32_bf16(false, a, false, bfr,
                                                  (short)0, acc, false, false);
  }
  const int lane = threadIdx.x & 31, half = lane >> 4;
  const int c = hh * HD + td + (lane & 15);
#pragma unroll
  for (int r = 0; r < 8; ++r) {
    const int n = tn + r + 8 * half;
    ret[(size_t)(b * SEQ + n) * DM + c] = f2bf(acc[r]);
  }
}

// ------------------------- y = silu(g) * ret -----------------------------------
__global__ void k_silu_mul(const unsigned short* __restrict__ g,
                           const unsigned short* __restrict__ ret,
                           unsigned short* __restrict__ y) {
  const size_t i = (size_t)blockIdx.x * 256 + threadIdx.x;
  const float gv = bf2f(g[i]);
  const float s = gv / (1.0f + __expf(-gv));
  y[i] = f2bf(s * bf2f(ret[i]));
}

// -------- FFN1: h[b,j] = gelu( sum_c yn[b,c] * W1[j,c] ); 8 j per block --------
__global__ void k_ffn1(const float* __restrict__ yn, const float* __restrict__ W1,
                       float* __restrict__ h) {
  const int tid = threadIdx.x;
  const int j0 = blockIdx.x * 8;
  float acc[8][8];
#pragma unroll
  for (int jj = 0; jj < 8; ++jj)
#pragma unroll
    for (int b = 0; b < 8; ++b) acc[jj][b] = 0.0f;

  for (int c = tid * 4; c < FLATD; c += 1024) {
    float4 wv[8];
#pragma unroll
    for (int jj = 0; jj < 8; ++jj)
      wv[jj] = *reinterpret_cast<const float4*>(W1 + (size_t)(j0 + jj) * FLATD + c);
#pragma unroll
    for (int b = 0; b < 8; ++b) {
      const float4 yv = *reinterpret_cast<const float4*>(yn + (size_t)b * FLATD + c);
#pragma unroll
      for (int jj = 0; jj < 8; ++jj)
        acc[jj][b] += wv[jj].x * yv.x + wv[jj].y * yv.y + wv[jj].z * yv.z + wv[jj].w * yv.w;
    }
  }

  __shared__ float red[64][8];
  const int lane = tid & 31, wave = tid >> 5;
#pragma unroll
  for (int jj = 0; jj < 8; ++jj)
#pragma unroll
    for (int b = 0; b < 8; ++b) {
      float r = acc[jj][b];
      for (int off = 16; off > 0; off >>= 1) r += __shfl_xor(r, off, 32);
      if (lane == 0) red[jj * 8 + b][wave] = r;
    }
  __syncthreads();
  if (tid < 64) {
    float s = 0.0f;
#pragma unroll
    for (int wv2 = 0; wv2 < 8; ++wv2) s += red[tid][wv2];
    const float ge = 0.5f * s * (1.0f + erff(s * 0.70710678f));  // exact gelu
    const int jj = tid >> 3, b = tid & 7;
    h[(size_t)b * FFNS + j0 + jj] = ge;
  }
}

// -------- FFN2: out[b,k] = Ysupl[b,k] + sum_j h[b,j] * W2[k,j] -----------------
__global__ void k_ffn2(const float* __restrict__ h, const float* __restrict__ W2,
                       const float* __restrict__ ysupl, float* __restrict__ out) {
  const int kk = blockIdx.x * 256 + threadIdx.x;
  const float4* w4 = reinterpret_cast<const float4*>(W2 + (size_t)kk * FFNS);
  float acc[8];
#pragma unroll
  for (int b = 0; b < 8; ++b) acc[b] = 0.0f;
  for (int j4 = 0; j4 < FFNS / 4; ++j4) {
    const float4 w = w4[j4];
#pragma unroll
    for (int b = 0; b < 8; ++b) {
      // block-uniform address -> scalar (s_load) path
      const float4 hv = *reinterpret_cast<const float4*>(h + (size_t)b * FFNS + j4 * 4);
      acc[b] += w.x * hv.x + w.y * hv.y + w.z * hv.z + w.w * hv.w;
    }
  }
#pragma unroll
  for (int b = 0; b < 8; ++b) {
    const size_t o = (size_t)b * FLATD + kk;
    out[o] = acc[b] + ysupl[o];
  }
}

extern "C" void kernel_launch(void* const* d_in, const int* in_sizes, int n_in,
                              void* d_out, int out_size, void* d_ws, size_t ws_size,
                              hipStream_t stream) {
  const float* X   = (const float*)d_in[0];
  const float* WQ  = (const float*)d_in[1];
  const float* WK  = (const float*)d_in[2];
  const float* WV  = (const float*)d_in[3];
  const float* WG  = (const float*)d_in[4];
  const float* WO  = (const float*)d_in[5];
  const float* lnw = (const float*)d_in[6];
  const float* lnb = (const float*)d_in[7];
  const float* W1  = (const float*)d_in[8];
  const float* W2  = (const float*)d_in[9];

  char* ws = (char*)d_ws;
  size_t off = 0;
  auto carve = [&](size_t bytes) { void* p = ws + off; off += (bytes + 255) & ~(size_t)255; return p; };
  unsigned short* xn   = (unsigned short*)carve((size_t)TOK * DM * 2);
  unsigned short* wqT  = (unsigned short*)carve((size_t)DM * DM * 2);
  unsigned short* wkT  = (unsigned short*)carve((size_t)DM * DM * 2);
  unsigned short* wvT  = (unsigned short*)carve((size_t)DM * DM * 2);
  unsigned short* wgT  = (unsigned short*)carve((size_t)DM * DM * 2);
  unsigned short* woT  = (unsigned short*)carve((size_t)DM * DM * 2);
  unsigned short* qb   = (unsigned short*)carve((size_t)TOK * DM * 2);
  unsigned short* kb   = (unsigned short*)carve((size_t)TOK * DM * 2);
  unsigned short* vb   = (unsigned short*)carve((size_t)TOK * DM * 2);
  unsigned short* gb   = (unsigned short*)carve((size_t)TOK * DM * 2);
  unsigned short* sc   = (unsigned short*)carve((size_t)BATCH * NH * SEQ * SEQ * 2);
  unsigned short* ret  = (unsigned short*)carve((size_t)TOK * DM * 2);
  unsigned short* yb   = (unsigned short*)carve((size_t)TOK * DM * 2);
  float* ysupl         = (float*)carve((size_t)TOK * DM * 4);
  float* yn            = (float*)carve((size_t)TOK * DM * 4);
  float* hb            = (float*)carve((size_t)BATCH * FFNS * 4);

  // 1) xn = LN(X) (bf16 for WMMA; f32 copy unused here, overwritten later)
  k_layernorm<<<TOK, 256, 0, stream>>>(X, lnw, lnb, xn, yn);
  // 2) weights -> bf16 transposed [N,K]
  k_transpose_bf16<<<(DM * DM) / 256, 256, 0, stream>>>(WQ, wqT);
  k_transpose_bf16<<<(DM * DM) / 256, 256, 0, stream>>>(WK, wkT);
  k_transpose_bf16<<<(DM * DM) / 256, 256, 0, stream>>>(WV, wvT);
  k_transpose_bf16<<<(DM * DM) / 256, 256, 0, stream>>>(WG, wgT);
  k_transpose_bf16<<<(DM * DM) / 256, 256, 0, stream>>>(WO, woT);
  // 3) Q/K/V/G projections (WMMA bf16): 2048x512x512 each; 4096 tiles / 8 waves
  k_gemm_bf16<<<512, 256, 0, stream>>>(xn, wqT, qb, nullptr, nullptr, TOK, DM, DM);
  k_gemm_bf16<<<512, 256, 0, stream>>>(xn, wkT, kb, nullptr, nullptr, TOK, DM, DM);
  k_gemm_bf16<<<512, 256, 0, stream>>>(xn, wvT, vb, nullptr, nullptr, TOK, DM, DM);
  k_gemm_bf16<<<512, 256, 0, stream>>>(xn, wgT, gb, nullptr, nullptr, TOK, DM, DM);
  // 4) decayed retention scores (WMMA) -> bf16
  k_scores<<<2048, 256, 0, stream>>>(qb, kb, sc);
  // 5) retention output = scores @ V (WMMA)
  k_retout<<<512, 256, 0, stream>>>(sc, vb, ret);
  // 6) y = silu(g) * ret
  k_silu_mul<<<(TOK * DM) / 256, 256, 0, stream>>>(gb, ret, yb);
  // 7) Ysupl = y @ W_O + X (WMMA, f32 out + residual)
  k_gemm_bf16<<<512, 256, 0, stream>>>(yb, woT, nullptr, ysupl, X, TOK, DM, DM);
  // 8) yn = LN(Ysupl) (f32 for FFN; bf16 copy goes to dead xn buffer)
  k_layernorm<<<TOK, 256, 0, stream>>>(ysupl, lnw, lnb, xn, yn);
  // 9) FFN (bandwidth-bound f32: stream W1/W2 exactly once)
  k_ffn1<<<FFNS / 8, 256, 0, stream>>>(yn, W1, hb);
  k_ffn2<<<FLATD / 256, 256, 0, stream>>>(hb, W2, ysupl, (float*)d_out);

  (void)in_sizes; (void)n_in; (void)out_size; (void)ws_size;
}
